// SAINR_41120016892687
// MI455X (gfx1250) — compile-verified, matrix operand order checked by reading, not run on hardware
//
#include <hip/hip_runtime.h>
#include <hip/hip_bf16.h>

typedef unsigned short u16;
typedef unsigned int   u32;
typedef __attribute__((ext_vector_type(16))) __bf16 bf16x16;
typedef __attribute__((ext_vector_type(8)))  float  f32x8;

#define DD 24
#define HH 96
#define WW 96
#define DHW (DD*HH*WW)
#define CC 64
#define KCONV (CC*27)       /* 1728, reordered as k = t*64 + ci */
#define LDAT 40             /* LDS row stride in shorts (16B-aligned) */

__device__ __forceinline__ u16 f2bf(float f) {
  union { float f; u32 u; } c; c.f = f;
  u32 u = c.u;
  u32 r = u + 0x7FFFu + ((u >> 16) & 1u);
  return (u16)(r >> 16);
}

union FragB16 {
  bf16x16 bf;
  u16  u[16];
  uint4 q[2];
};

union Stage8 {
  u16  s[8];
  uint4 q;
};

// Build a 16x32 bf16 WMMA fragment from an LDS tile (row-major, stride LDAT).
// Per CDNA5 ISA 16-bit A layout: lanes 0-15 -> row, K{0..7,16..23};
// lanes 16-31 -> row, K{8..15,24..31}. B^T tiles use the same gather.
__device__ __forceinline__ FragB16 ldsFrag(const u16* lds, int rowBase, int lane) {
  FragB16 f;
  int r   = rowBase + (lane & 15);
  int klo = (lane < 16) ? 0 : 8;
  const u16* p = lds + r * LDAT + klo;
  f.q[0] = *(const uint4*)(p);
  f.q[1] = *(const uint4*)(p + 16);
  return f;
}

// ---------------------------------------------------------------------------
// Implicit-GEMM 3x3x3 conv, Cin=Cout=64, SAME pad, via v_wmma_f32_16x16x32_bf16
// Block tile: M=32 (contiguous W positions), N=64 (all Cout). 128 thr = 4 waves.
// K ordering is tap-major (k = t*64 + ci): tap decode + (zd,zh) bounds are
// uniform per K step (SALU). A staging: lane -> m (coalesced w), wave -> 8
// consecutive ci; loads are clamped+cndmask (no EXEC branches), LDS writes are
// one ds_store_b128 per thread.
// ---------------------------------------------------------------------------
__global__ __launch_bounds__(128)
void conv64_wmma(const float* __restrict__ src, const u16* __restrict__ wbf,
                 const float* __restrict__ bias, const float* __restrict__ addsrc,
                 float* __restrict__ out, int relu)
{
  __shared__ __align__(16) u16 As[32 * LDAT];
  __shared__ __align__(16) u16 Bs[64 * LDAT];

  int bid = blockIdx.x;
  int tw  = bid % 3;
  int h   = (bid / 3) % HH;
  int d   = (bid / (3 * HH)) % DD;
  int n   = bid / (3 * HH * DD);
  int w0  = tw * 32;

  int tid  = threadIdx.x;
  int lane = tid & 31;
  int wave = tid >> 5;
  int coBase = wave * 16;

  f32x8 acc0 = {};
  f32x8 acc1 = {};

  // B staging: each thread copies 16 shorts of one Cout row.
  int brow = tid >> 1, bkq = (tid & 1) * 16;

  const float* srcN = src + (size_t)n * CC * DHW;

  for (int t = 0; t < 27; ++t) {
    int kd  = t / 9;
    int r9  = t - kd * 9;
    int kh3 = r9 / 3;
    int kw  = r9 - kh3 * 3;
    int zd  = d + kd - 1;
    int zh  = h + kh3 - 1;
    bool rowOK = ((unsigned)zd < (unsigned)DD) && ((unsigned)zh < (unsigned)HH);
    int zdc = min(max(zd, 0), DD - 1);
    int zhc = min(max(zh, 0), HH - 1);
    const float* srcRowC = srcN + (size_t)(zdc * HH + zhc) * WW;

    int zw   = w0 + lane + kw - 1;           // lane == m row
    bool okw = rowOK && ((unsigned)zw < (unsigned)WW);
    int zwc  = min(max(zw, 0), WW - 1);

    #pragma unroll
    for (int half = 0; half < 2; ++half) {
      int kb  = t * 64 + half * 32;
      int ciB = half * 32 + wave * 8;
      // A: 8 coalesced loads (lane-contiguous w), clamp+select, 1 b128 LDS store
      Stage8 stg;
      #pragma unroll
      for (int e = 0; e < 8; ++e) {
        float v = srcRowC[(size_t)(ciB + e) * DHW + zwc];
        stg.s[e] = f2bf(okw ? v : 0.f);
      }
      *(uint4*)&As[lane * LDAT + wave * 8] = stg.q;
      // B: contiguous copy from packed bf16 weights [Cout][1728]
      {
        const u16* wp = wbf + (size_t)brow * KCONV + kb + bkq;
        *(uint4*)&Bs[brow * LDAT + bkq]     = *(const uint4*)wp;
        *(uint4*)&Bs[brow * LDAT + bkq + 8] = *(const uint4*)(wp + 8);
        __builtin_prefetch(wp + 32, 0, 0);   // next K slice -> global_prefetch_b8
      }
      __syncthreads();

      FragB16 a0 = ldsFrag(As, 0,  lane);
      FragB16 a1 = ldsFrag(As, 16, lane);
      FragB16 b  = ldsFrag(Bs, coBase, lane);
      acc0 = __builtin_amdgcn_wmma_f32_16x16x32_bf16(false, a0.bf, false, b.bf,
                                                     (short)0, acc0, false, false);
      acc1 = __builtin_amdgcn_wmma_f32_16x16x32_bf16(false, a1.bf, false, b.bf,
                                                     (short)0, acc1, false, false);
      __syncthreads();
    }
  }

  // Epilogue per CDNA5 C/D layout: VGPR r -> (M = r + (lane>=16 ? 8 : 0), N = lane&15)
  // Each lane's 8 outputs are w-contiguous -> 2x float4 per accumulator.
  int co   = coBase + (lane & 15);
  int mAdd = (lane < 16) ? 0 : 8;
  float bv = bias ? bias[co] : 0.f;
  size_t obase = (size_t)(n * CC + co) * DHW + (d * HH + h) * WW + w0;
  #pragma unroll
  for (int rr = 0; rr < 8; rr += 4) {
    float4 v;
    v.x = acc0[rr + 0] + bv; v.y = acc0[rr + 1] + bv;
    v.z = acc0[rr + 2] + bv; v.w = acc0[rr + 3] + bv;
    if (relu) { v.x = fmaxf(v.x, 0.f); v.y = fmaxf(v.y, 0.f);
                v.z = fmaxf(v.z, 0.f); v.w = fmaxf(v.w, 0.f); }
    size_t idx = obase + mAdd + rr;
    if (addsrc) { float4 a = *(const float4*)&addsrc[idx];
                  v.x += a.x; v.y += a.y; v.z += a.z; v.w += a.w; }
    *(float4*)&out[idx] = v;
  }
  #pragma unroll
  for (int rr = 0; rr < 8; rr += 4) {
    float4 v;
    v.x = acc1[rr + 0] + bv; v.y = acc1[rr + 1] + bv;
    v.z = acc1[rr + 2] + bv; v.w = acc1[rr + 3] + bv;
    if (relu) { v.x = fmaxf(v.x, 0.f); v.y = fmaxf(v.y, 0.f);
                v.z = fmaxf(v.z, 0.f); v.w = fmaxf(v.w, 0.f); }
    size_t idx = obase + 16 + mAdd + rr;
    if (addsrc) { float4 a = *(const float4*)&addsrc[idx];
                  v.x += a.x; v.y += a.y; v.z += a.z; v.w += a.w; }
    *(float4*)&out[idx] = v;
  }
}

// ---------------------------------------------------------------------------
// Generic GEMM: Out[M,N] = A[M,K](f32) * B^T[N,K](bf16) + bias, optional ReLU.
// K % 32 == 0, N % 64 == 0. Same tiling as conv kernel.
// ---------------------------------------------------------------------------
__global__ __launch_bounds__(128)
void gemm_wmma(const float* __restrict__ A, const u16* __restrict__ Bt,
               const float* __restrict__ bias, float* __restrict__ Out,
               int M, int K, int N, int relu)
{
  __shared__ __align__(16) u16 As[32 * LDAT];
  __shared__ __align__(16) u16 Bs[64 * LDAT];

  int mBase = blockIdx.x * 32;
  int nBase = blockIdx.y * 64;
  int tid = threadIdx.x, lane = tid & 31, wave = tid >> 5;
  int coBase = wave * 16;

  f32x8 acc0 = {};
  f32x8 acc1 = {};

  int am  = tid >> 2, akq = (tid & 3) * 8;
  int brow = tid >> 1, bkq = (tid & 1) * 16;
  int row = mBase + am;

  for (int kb = 0; kb < K; kb += 32) {
    // A: 2x global_load_b128 (f32) -> bf16 -> 1x ds_store_b128
    float4 f0 = {0.f, 0.f, 0.f, 0.f}, f1 = {0.f, 0.f, 0.f, 0.f};
    if (row < M) {
      const float4* ap = (const float4*)(A + (size_t)row * K + kb + akq);
      f0 = ap[0]; f1 = ap[1];
    }
    Stage8 stg;
    stg.s[0] = f2bf(f0.x); stg.s[1] = f2bf(f0.y); stg.s[2] = f2bf(f0.z); stg.s[3] = f2bf(f0.w);
    stg.s[4] = f2bf(f1.x); stg.s[5] = f2bf(f1.y); stg.s[6] = f2bf(f1.z); stg.s[7] = f2bf(f1.w);
    *(uint4*)&As[am * LDAT + akq] = stg.q;
    {
      const u16* wp = Bt + (size_t)(nBase + brow) * K + kb + bkq;
      *(uint4*)&Bs[brow * LDAT + bkq]     = *(const uint4*)wp;
      *(uint4*)&Bs[brow * LDAT + bkq + 8] = *(const uint4*)(wp + 8);
      __builtin_prefetch(wp + 32, 0, 0);
    }
    __syncthreads();
    FragB16 a0 = ldsFrag(As, 0,  lane);
    FragB16 a1 = ldsFrag(As, 16, lane);
    FragB16 b  = ldsFrag(Bs, coBase, lane);
    acc0 = __builtin_amdgcn_wmma_f32_16x16x32_bf16(false, a0.bf, false, b.bf,
                                                   (short)0, acc0, false, false);
    acc1 = __builtin_amdgcn_wmma_f32_16x16x32_bf16(false, a1.bf, false, b.bf,
                                                   (short)0, acc1, false, false);
    __syncthreads();
  }

  int col  = nBase + coBase + (lane & 15);
  int mAdd = (lane < 16) ? 0 : 8;
  float bv = bias ? bias[col] : 0.f;
  #pragma unroll
  for (int r = 0; r < 8; ++r) {
    int m = mBase + r + mAdd;
    if (m < M) {
      float v = acc0[r] + bv;
      if (relu) v = fmaxf(v, 0.f);
      Out[(size_t)m * N + col] = v;
    }
  }
  #pragma unroll
  for (int r = 0; r < 8; ++r) {
    int m = mBase + 16 + r + mAdd;
    if (m < M) {
      float v = acc1[r] + bv;
      if (relu) v = fmaxf(v, 0.f);
      Out[(size_t)m * N + col] = v;
    }
  }
}

// ---------------------------------------------------------------------------
// Weight packing
// ---------------------------------------------------------------------------
// conv weights: in f32 [co][ci][27] -> out bf16 [co][t*64 + ci]
__global__ void pack_conv_weights(const float* __restrict__ in, u16* __restrict__ out) {
  int i = blockIdx.x * blockDim.x + threadIdx.x;
  if (i >= CC * KCONV) return;
  int co = i / KCONV;
  int r  = i - co * KCONV;
  int t  = r / CC;
  int ci = r - t * CC;
  out[i] = f2bf(in[(co * CC + ci) * 27 + t]);
}

// dense weights: in f32 [K][N] row-major -> out bf16 [N][K]
__global__ void pack_transpose(const float* __restrict__ in, u16* __restrict__ out, int K, int N) {
  int i = blockIdx.x * blockDim.x + threadIdx.x;
  if (i >= K * N) return;
  int n = i / K, k = i - n * K;
  out[(size_t)n * K + k] = f2bf(in[(size_t)k * N + n]);
}

// ---------------------------------------------------------------------------
// Scalar helper kernels
// ---------------------------------------------------------------------------
__global__ void head_conv(const float* __restrict__ inp, const float* __restrict__ hw,
                          const float* __restrict__ hb, float* __restrict__ out, int total) {
  int id = blockIdx.x * blockDim.x + threadIdx.x;
  if (id >= total) return;
  int s = id % DHW;
  int c = (id / DHW) % CC;
  int n = id / (DHW * CC);
  int w = s % WW, hh = (s / WW) % HH, dd = s / (WW * HH);
  float acc = hb[c];
  #pragma unroll
  for (int kd = 0; kd < 3; ++kd)
    #pragma unroll
    for (int kh = 0; kh < 3; ++kh)
      #pragma unroll
      for (int kw = 0; kw < 3; ++kw) {
        int zd = dd + kd - 1, zh = hh + kh - 1, zw = w + kw - 1;
        if (zd >= 0 && zd < DD && zh >= 0 && zh < HH && zw >= 0 && zw < WW)
          acc += hw[c * 27 + kd * 9 + kh * 3 + kw] *
                 inp[(size_t)n * DHW + (zd * HH + zh) * WW + zw];
      }
  out[id] = acc;
}

__device__ __forceinline__ float tri_sample(const float* __restrict__ vol, int b, int c,
                                            float cd, float ch, float cw) {
  float d = (cd + 1.f) * 0.5f * (DD - 1);
  float h = (ch + 1.f) * 0.5f * (HH - 1);
  float w = (cw + 1.f) * 0.5f * (WW - 1);
  float d0f = floorf(d), h0f = floorf(h), w0f = floorf(w);
  float fd = d - d0f, fh = h - h0f, fw = w - w0f;
  int d0 = (int)d0f, h0 = (int)h0f, w0 = (int)w0f;
  const float* base = vol + (size_t)(b * CC + c) * DHW;
  float acc = 0.f;
  #pragma unroll
  for (int dz = 0; dz < 2; ++dz)
    #pragma unroll
    for (int dy = 0; dy < 2; ++dy)
      #pragma unroll
      for (int dx = 0; dx < 2; ++dx) {
        int di = d0 + dz, hi = h0 + dy, wi = w0 + dx;
        float wt = (dz ? fd : 1.f - fd) * (dy ? fh : 1.f - fh) * (dx ? fw : 1.f - fw);
        bool inb = (di >= 0) && (di < DD) && (hi >= 0) && (hi < HH) && (wi >= 0) && (wi < WW);
        int dc = min(max(di, 0), DD - 1), hc = min(max(hi, 0), HH - 1), wc = min(max(wi, 0), WW - 1);
        acc += base[(dc * HH + hc) * WW + wc] * (inb ? wt : 0.f);
      }
  return acc;
}

__global__ void sample_points(const float* __restrict__ feat, const float* __restrict__ coord,
                              float* __restrict__ out, int npts) {
  int id = blockIdx.x * blockDim.x + threadIdx.x;
  if (id >= npts * CC) return;
  int c = id % CC, p = id / CC, b = p / 4096;
  out[id] = tri_sample(feat, b, c, coord[p * 3 + 0], coord[p * 3 + 1], coord[p * 3 + 2]);
}

__global__ void sample_neighbors(const float* __restrict__ feat, const float* __restrict__ proj,
                                 float* __restrict__ out, int total) {
  int id = blockIdx.x * blockDim.x + threadIdx.x;
  if (id >= total) return;
  int c = id % CC;
  int loc = id / CC;
  int kk = loc % 98;
  int p = loc / 98;
  int b = p / 4096;
  int kd = kk / 49, rem = kk % 49, kh = rem / 7, kw = rem % 7;
  float od = ((float)kd - 0.5f) * (2.f / (DD - 1));
  float oh = ((float)kh - 3.0f) * (2.f / (HH - 1));
  float ow = ((float)kw - 3.0f) * (2.f / (WW - 1));
  out[id] = tri_sample(feat, b, c,
                       proj[p * 3 + 0] + od, proj[p * 3 + 1] + oh, proj[p * 3 + 2] + ow);
}

__global__ __launch_bounds__(128)
void attn_combine(const float* __restrict__ qmat, const float* __restrict__ kmat,
                  const float* __restrict__ vmat, const float* __restrict__ qfeat,
                  const float* __restrict__ wo, float* __restrict__ out) {
  __shared__ float sl[98];
  __shared__ float so[64];
  __shared__ float sred;
  int p = blockIdx.x, tid = threadIdx.x;
  if (tid < 98) {
    const float* q  = qmat + (size_t)p * 64;
    const float* kr = kmat + ((size_t)p * 98 + tid) * 64;
    float acc = 0.f;
    for (int c = 0; c < 64; ++c) acc += q[c] * kr[c];
    sl[tid] = acc * 0.125f;    // 1/sqrt(64)
  }
  __syncthreads();
  if (tid == 0) { float m = sl[0]; for (int i = 1; i < 98; ++i) m = fmaxf(m, sl[i]); sred = m; }
  __syncthreads();
  if (tid < 98) sl[tid] = expf(sl[tid] - sred);
  __syncthreads();
  if (tid == 0) { float s = 0.f; for (int i = 0; i < 98; ++i) s += sl[i]; sred = s; }
  __syncthreads();
  if (tid < 98) sl[tid] /= sred;
  __syncthreads();
  if (tid < 64) {
    float acc = 0.f;
    for (int kk = 0; kk < 98; ++kk) acc += sl[kk] * vmat[((size_t)p * 98 + kk) * 64 + tid];
    so[tid] = acc;
  }
  __syncthreads();
  if (tid < 64) {
    float r = qfeat[(size_t)p * 64 + tid];
    for (int c = 0; c < 64; ++c) r += so[c] * wo[c * 64 + tid];
    out[(size_t)p * 64 + tid] = r;
  }
}

__global__ void mlp_last(const float* __restrict__ in, const float* __restrict__ w,
                         const float* __restrict__ b, float* __restrict__ out, int npts, int K) {
  int p = blockIdx.x * blockDim.x + threadIdx.x;
  if (p >= npts) return;
  float acc = b[0];
  for (int k = 0; k < K; ++k) acc += in[(size_t)p * K + k] * w[k];
  out[p] = acc;
}

__global__ void final_combine(const float* __restrict__ qfeat, const float* __restrict__ mask_w,
                              const float* __restrict__ mask_b, const float* __restrict__ gum,
                              const float* __restrict__ easy, const float* __restrict__ diff,
                              float* __restrict__ out, int npts) {
  int p = blockIdx.x * blockDim.x + threadIdx.x;
  if (p >= npts) return;
  float l0 = mask_b[0], l1 = mask_b[1];
  for (int c = 0; c < 64; ++c) {
    float q = qfeat[(size_t)p * 64 + c];
    l0 += q * mask_w[c * 2 + 0];
    l1 += q * mask_w[c * 2 + 1];
  }
  float g0 = -logf(-logf(gum[p * 2 + 0] + 1e-10f) + 1e-10f);
  float g1 = -logf(-logf(gum[p * 2 + 1] + 1e-10f) + 1e-10f);
  float m1 = ((l1 + g1) > (l0 + g0)) ? 1.f : 0.f;   // hard one-hot (fwd value of ST-gumbel)
  out[p] = (1.f - m1) * easy[p] + m1 * diff[p];
  out[npts + p] = m1;
}

// ---------------------------------------------------------------------------
extern "C" void kernel_launch(void* const* d_in, const int* in_sizes, int n_in,
                              void* d_out, int out_size, void* d_ws, size_t ws_size,
                              hipStream_t stream) {
  (void)in_sizes; (void)n_in; (void)out_size; (void)ws_size;
  const float* inp      = (const float*)d_in[0];
  const float* hr_coord = (const float*)d_in[1];
  const float* proj     = (const float*)d_in[2];
  const float* gum      = (const float*)d_in[3];
  const float* head_w   = (const float*)d_in[4];
  const float* head_b   = (const float*)d_in[5];
  const float* tail_w   = (const float*)d_in[38];
  const float* tail_b   = (const float*)d_in[39];
  const float* mask_w   = (const float*)d_in[40];
  const float* mask_b   = (const float*)d_in[41];
  const float* wq = (const float*)d_in[42];
  const float* wk = (const float*)d_in[43];
  const float* wv = (const float*)d_in[44];
  const float* wo = (const float*)d_in[45];

  const int NP = 2 * 4096;
  const size_t VOL = (size_t)2 * CC * DHW;

  char* ws = (char*)d_ws;
  size_t off = 0;
  auto alloc = [&](size_t bytes) -> void* {
    void* p = ws + off;
    off = (off + bytes + 255) & ~(size_t)255;
    return p;
  };

  float* xbuf = (float*)alloc(VOL * 4);
  float* bufA = (float*)alloc(VOL * 4);
  float* bufB = (float*)alloc(VOL * 4);
  float* tmp  = (float*)alloc(VOL * 4);
  float* feat = (float*)alloc(VOL * 4);
  const int wElems = CC * KCONV;
  u16* wconv = (u16*)alloc((size_t)17 * wElems * 2);
  float* qfeat = (float*)alloc((size_t)NP * 64 * 4);
  float* qmat  = (float*)alloc((size_t)NP * 64 * 4);
  float* nbf   = (float*)alloc((size_t)NP * 98 * 64 * 4);
  float* kmat  = (float*)alloc((size_t)NP * 98 * 64 * 4);
  float* vmat  = (float*)alloc((size_t)NP * 98 * 64 * 4);
  float* attn  = (float*)alloc((size_t)NP * 64 * 4);
  float* mA    = (float*)alloc((size_t)NP * 256 * 4);
  float* mB    = (float*)alloc((size_t)NP * 256 * 4);
  float* peasy = (float*)alloc(NP * 4);
  float* pdiff = (float*)alloc(NP * 4);
  u16* wqT = (u16*)alloc(64 * 64 * 2);
  u16* wkT = (u16*)alloc(64 * 64 * 2);
  u16* wvT = (u16*)alloc(64 * 64 * 2);
  u16* m0T = (u16*)alloc(256 * 64 * 2);
  u16* m1T = (u16*)alloc(256 * 256 * 2);
  u16* m2T = (u16*)alloc(256 * 256 * 2);
  u16* m3T = (u16*)alloc(256 * 256 * 2);

  // Pack conv weights to bf16 [Cout][t*64+ci] (tap-major K, = B^T for implicit GEMM)
  int cg = (wElems + 255) / 256;
  for (int i = 0; i < 8; ++i) {
    pack_conv_weights<<<cg, 256, 0, stream>>>((const float*)d_in[6 + 4 * i],
                                              wconv + (size_t)(2 * i) * wElems);
    pack_conv_weights<<<cg, 256, 0, stream>>>((const float*)d_in[8 + 4 * i],
                                              wconv + (size_t)(2 * i + 1) * wElems);
  }
  pack_conv_weights<<<cg, 256, 0, stream>>>(tail_w, wconv + (size_t)16 * wElems);

  // Pack dense weights transposed to bf16 [N][K]
  pack_transpose<<<(64 * 64 + 255) / 256, 256, 0, stream>>>(wq, wqT, 64, 64);
  pack_transpose<<<(64 * 64 + 255) / 256, 256, 0, stream>>>(wk, wkT, 64, 64);
  pack_transpose<<<(64 * 64 + 255) / 256, 256, 0, stream>>>(wv, wvT, 64, 64);
  pack_transpose<<<(64 * 256 + 255) / 256, 256, 0, stream>>>((const float*)d_in[46], m0T, 64, 256);
  pack_transpose<<<(256 * 256 + 255) / 256, 256, 0, stream>>>((const float*)d_in[48], m1T, 256, 256);
  pack_transpose<<<(256 * 256 + 255) / 256, 256, 0, stream>>>((const float*)d_in[50], m2T, 256, 256);
  pack_transpose<<<(256 * 256 + 255) / 256, 256, 0, stream>>>((const float*)d_in[52], m3T, 256, 256);

  // Head conv (Cin=1, scalar) -> x
  int totalHead = (int)VOL;
  head_conv<<<(totalHead + 255) / 256, 256, 0, stream>>>(inp, head_w, head_b, xbuf, totalHead);

  // Residual conv stack via WMMA implicit GEMM
  const int convGrid = 2 * DD * HH * 3;   // 13824 blocks, M-tile = 32 along W
  const float* cur = xbuf;
  float* pp[2] = {bufA, bufB};
  for (int i = 0; i < 8; ++i) {
    conv64_wmma<<<convGrid, 128, 0, stream>>>(cur, wconv + (size_t)(2 * i) * wElems,
                                              (const float*)d_in[7 + 4 * i], nullptr, tmp, 1);
    float* nxt = pp[i & 1];
    conv64_wmma<<<convGrid, 128, 0, stream>>>(tmp, wconv + (size_t)(2 * i + 1) * wElems,
                                              (const float*)d_in[9 + 4 * i], cur, nxt, 0);
    cur = nxt;
  }
  // feat = tail(res) + x
  conv64_wmma<<<convGrid, 128, 0, stream>>>(cur, wconv + (size_t)16 * wElems,
                                            tail_b, xbuf, feat, 0);

  // q_feat = trilinear(feat, hr_coord)
  sample_points<<<(NP * 64 + 255) / 256, 256, 0, stream>>>(feat, hr_coord, qfeat, NP);

  // Attention branch
  gemm_wmma<<<dim3((NP + 31) / 32, 1), 128, 0, stream>>>(qfeat, wqT, nullptr, qmat, NP, 64, 64, 0);
  int totNb = NP * 98 * 64;
  sample_neighbors<<<(totNb + 255) / 256, 256, 0, stream>>>(feat, proj, nbf, totNb);
  int Mnb = NP * 98;
  gemm_wmma<<<dim3((Mnb + 31) / 32, 1), 128, 0, stream>>>(nbf, wkT, nullptr, kmat, Mnb, 64, 64, 0);
  gemm_wmma<<<dim3((Mnb + 31) / 32, 1), 128, 0, stream>>>(nbf, wvT, nullptr, vmat, Mnb, 64, 64, 0);
  attn_combine<<<NP, 128, 0, stream>>>(qmat, kmat, vmat, qfeat, wo, attn);

  // MLP (easy branch): q_feat -> pred_easy
  gemm_wmma<<<dim3((NP + 31) / 32, 4), 128, 0, stream>>>(qfeat, m0T, (const float*)d_in[47], mA, NP, 64, 256, 1);
  gemm_wmma<<<dim3((NP + 31) / 32, 4), 128, 0, stream>>>(mA, m1T, (const float*)d_in[49], mB, NP, 256, 256, 1);
  gemm_wmma<<<dim3((NP + 31) / 32, 4), 128, 0, stream>>>(mB, m2T, (const float*)d_in[51], mA, NP, 256, 256, 1);
  gemm_wmma<<<dim3((NP + 31) / 32, 4), 128, 0, stream>>>(mA, m3T, (const float*)d_in[53], mB, NP, 256, 256, 1);
  mlp_last<<<(NP + 255) / 256, 256, 0, stream>>>(mB, (const float*)d_in[54], (const float*)d_in[55], peasy, NP, 256);

  // MLP (diff branch): attn -> pred_diff
  gemm_wmma<<<dim3((NP + 31) / 32, 4), 128, 0, stream>>>(attn, m0T, (const float*)d_in[47], mA, NP, 64, 256, 1);
  gemm_wmma<<<dim3((NP + 31) / 32, 4), 128, 0, stream>>>(mA, m1T, (const float*)d_in[49], mB, NP, 256, 256, 1);
  gemm_wmma<<<dim3((NP + 31) / 32, 4), 128, 0, stream>>>(mB, m2T, (const float*)d_in[51], mA, NP, 256, 256, 1);
  gemm_wmma<<<dim3((NP + 31) / 32, 4), 128, 0, stream>>>(mA, m3T, (const float*)d_in[53], mB, NP, 256, 256, 1);
  mlp_last<<<(NP + 255) / 256, 256, 0, stream>>>(mB, (const float*)d_in[54], (const float*)d_in[55], pdiff, NP, 256);

  // Gumbel hard mask + combine -> d_out[0:NP] = pred, d_out[NP:2NP] = mask[...,1]
  final_combine<<<(NP + 255) / 256, 256, 0, stream>>>(qfeat, mask_w, mask_b, gum,
                                                      peasy, pdiff, (float*)d_out, NP);
}
